// RNNModel_4870492913815
// MI455X (gfx1250) — matrix operand order repaired and checked by reading, not compiled, not run on hardware
//
#include <hip/hip_runtime.h>
#include <math.h>

typedef __attribute__((ext_vector_type(16))) __bf16 v16bf;
typedef __attribute__((ext_vector_type(8)))  float  v8f;
typedef __attribute__((ext_vector_type(4)))  float  v4f;

namespace {
constexpr int Bn = 512, Tn = 1024, Dn = 64, Hn = 64, Cn = 64;
constexpr int LDSS = 68;  // padded LDS row stride (floats) to break bank conflicts
}

union ABu { v16bf v; __bf16 e[16]; };
union CDu { v8f   v; float  e[8];  };

// ---- fragment helpers (layouts per CDNA5 ISA §7.12.2) ----

// B fragment: B(K,N) = W[N][K], W row-major [64][64]. Lane n = (lane&15)+16*nt,
// e=0..7 -> K = kt*32 + 8*hi + e ; e=8..15 -> K = kt*32 + 16 + 8*hi + (e-8)
__device__ __forceinline__ v16bf load_bfrag(const float* W, int n, int kt, int hi) {
  const float* p = W + n * 64 + kt * 32 + hi * 8;
  ABu r;
#pragma unroll
  for (int i = 0; i < 8; ++i) r.e[i] = (__bf16)p[i];
#pragma unroll
  for (int i = 0; i < 8; ++i) r.e[8 + i] = (__bf16)p[16 + i];
  return r.v;
}

// A fragment: row = pointer to 64 contiguous K-floats of row m = lane&15.
// e=0..7 -> K = kt*32 + 8*hi + e ; e=8..15 -> K = kt*32 + 16 + 8*hi + (e-8)
__device__ __forceinline__ v16bf load_afrag(const float* row, int kt, int hi) {
  const float* p = row + kt * 32 + hi * 8;
  ABu r;
#pragma unroll
  for (int i = 0; i < 8; ++i) r.e[i] = (__bf16)p[i];
#pragma unroll
  for (int i = 0; i < 8; ++i) r.e[8 + i] = (__bf16)p[16 + i];
  return r.v;
}

// Nontemporal A-fragment load (read-once streams: keep L2 for xW0)
__device__ __forceinline__ v16bf load_afrag_nt(const float* row, int kt, int hi) {
  const v4f* p = (const v4f*)(row + kt * 32 + hi * 8);
  v4f x0 = __builtin_nontemporal_load(p);
  v4f x1 = __builtin_nontemporal_load(p + 1);
  v4f x2 = __builtin_nontemporal_load(p + 4);  // +16 floats
  v4f x3 = __builtin_nontemporal_load(p + 5);
  ABu r;
#pragma unroll
  for (int i = 0; i < 4; ++i) {
    r.e[i]      = (__bf16)x0[i];
    r.e[4 + i]  = (__bf16)x1[i];
    r.e[8 + i]  = (__bf16)x2[i];
    r.e[12 + i] = (__bf16)x3[i];
  }
  return r.v;
}

__device__ __forceinline__ v8f wmma_bf16(v16bf a, v16bf b, v8f c) {
  return __builtin_amdgcn_wmma_f32_16x16x32_bf16(false, a, false, b, (short)0, c,
                                                 false, false);
}

__device__ __forceinline__ float fast_tanh(float x) {
#if __has_builtin(__builtin_amdgcn_tanhf)
  return __builtin_amdgcn_tanhf(x);
#else
  return tanhf(x);
#endif
}

__device__ __forceinline__ v8f splat8(float s) {
  CDu c;
#pragma unroll
  for (int v = 0; v < 8; ++v) c.e[v] = s;
  return c.v;
}

// ---------------- Kernel 1: input projection  xW0 = x @ Wih0.T + (bih0+bhh0)
// Output stored fragment-major: chunk id = ((t*32 + bt)*4 + nt), chunk = 256
// floats laid out as [lane][v] so the scan loads one v8f per lane per tile.
__global__ __launch_bounds__(256) void rnn_proj0_kernel(
    const float* __restrict__ x, const float* __restrict__ Wih0,
    const float* __restrict__ bih0, const float* __restrict__ bhh0,
    float* __restrict__ xw0) {
  const int lane = threadIdx.x & 31;
  const int wave = threadIdx.x >> 5;
  const int g = blockIdx.x * 8 + wave;  // tile id: t*32 + bt, 0..32767
  const int t = g >> 5;
  const int b0 = (g & 31) << 4;
  const int lm = lane & 15, hi = lane >> 4;

  v16bf Bf[2][4];
#pragma unroll
  for (int kt = 0; kt < 2; ++kt)
#pragma unroll
    for (int nt = 0; nt < 4; ++nt)
      Bf[kt][nt] = load_bfrag(Wih0, nt * 16 + lm, kt, hi);

  const float* row = x + ((size_t)(b0 + lm) * Tn + t) * Dn;
  v16bf a0 = load_afrag_nt(row, 0, hi);
  v16bf a1 = load_afrag_nt(row, 1, hi);

#pragma unroll
  for (int nt = 0; nt < 4; ++nt) {
    v8f c = splat8(bih0[nt * 16 + lm] + bhh0[nt * 16 + lm]);
    c = wmma_bf16(a0, Bf[0][nt], c);
    c = wmma_bf16(a1, Bf[1][nt], c);
    float* outp = xw0 + ((size_t)g * 4 + nt) * 256 + lane * 8;
    *(v8f*)outp = c;  // temporal store: xW0 (128 MB) should stay in 192 MB L2
  }
}

// ---------------- Kernel 2: fused scan (layer0 + layer1 + FC), one wave per
// 16 batch rows; 1024 sequential steps; transpose via LDS between stages.
// A-fragments rotate across iterations; xW0 tiles are double-buffered in
// registers with a deep global_prefetch ahead of them.
__global__ __launch_bounds__(32) void rnn_scan_fused_kernel(
    const float* __restrict__ xw0, const float* __restrict__ h0in,
    const float* __restrict__ Whh0, const float* __restrict__ Wih1,
    const float* __restrict__ Whh1, const float* __restrict__ bih1,
    const float* __restrict__ bhh1, const float* __restrict__ Wfc,
    const float* __restrict__ bfc, float* __restrict__ out,
    float* __restrict__ hout) {
  __shared__ float hs0[16 * LDSS];
  __shared__ float hs1[16 * LDSS];

  const int lane = threadIdx.x;
  const int lm = lane & 15, hi = lane >> 4;
  const int bt = blockIdx.x;  // 0..31
  const int b0 = bt << 4;

  // loop-invariant weight fragments held in VGPRs
  v16bf Bh0[2][4], Bi1[2][4], Bh1[2][4], Bfc_[2][4];
#pragma unroll
  for (int kt = 0; kt < 2; ++kt)
#pragma unroll
    for (int nt = 0; nt < 4; ++nt) {
      Bh0[kt][nt] = load_bfrag(Whh0, nt * 16 + lm, kt, hi);
      Bi1[kt][nt] = load_bfrag(Wih1, nt * 16 + lm, kt, hi);
      Bh1[kt][nt] = load_bfrag(Whh1, nt * 16 + lm, kt, hi);
      Bfc_[kt][nt] = load_bfrag(Wfc, nt * 16 + lm, kt, hi);
    }
  // pre-splatted loop-invariant C-accumulator seeds (D != C, no per-step movs)
  v8f biasC1[4], biasCf[4];
#pragma unroll
  for (int nt = 0; nt < 4; ++nt) {
    biasC1[nt] = splat8(bih1[nt * 16 + lm] + bhh1[nt * 16 + lm]);
    biasCf[nt] = splat8(bfc[nt * 16 + lm]);
  }
  const v8f zeroC = splat8(0.0f);

  // init h tiles in LDS from h0 [L,B,H]
  for (int i = lane; i < 16 * 64; i += 32) {
    int m = i >> 6, k = i & 63;
    hs0[m * LDSS + k] = h0in[(size_t)(b0 + m) * Hn + k];
    hs1[m * LDSS + k] = h0in[(size_t)Bn * Hn + (size_t)(b0 + m) * Hn + k];
  }
  __syncthreads();

  // rotating A-fragments: a0k* = h0(t-1), ahk* = h1(t-1)
  v16bf a0k0 = load_afrag(hs0 + lm * LDSS, 0, hi);
  v16bf a0k1 = load_afrag(hs0 + lm * LDSS, 1, hi);
  v16bf ahk0 = load_afrag(hs1 + lm * LDSS, 0, hi);
  v16bf ahk1 = load_afrag(hs1 + lm * LDSS, 1, hi);

  // software-pipelined xW0 tile (t=0)
  v8f xcur[4];
  {
    const float* xb = xw0 + (size_t)bt * 1024 + lane * 8;
#pragma unroll
    for (int nt = 0; nt < 4; ++nt) xcur[nt] = *(const v8f*)(xb + nt * 256);
  }

  v8f hn0[4], hn1[4];
#pragma unroll
  for (int nt = 0; nt < 4; ++nt) { hn0[nt] = zeroC; hn1[nt] = zeroC; }

#pragma unroll 1
  for (int t = 0; t < Tn; ++t) {
    // deep prefetch: 4 steps ahead, one lane-strided global_prefetch per tile
    if (t + 4 < Tn) {
      const char* pf =
          (const char*)(xw0 + ((size_t)(t + 4) * 32 + bt) * 1024) + lane * 128;
      __builtin_prefetch(pf, 0, 3);
    }
    // register prefetch: next step's xW0 tile
    v8f xnext[4];
    if (t + 1 < Tn) {
      const float* xb = xw0 + ((size_t)(t + 1) * 32 + bt) * 1024 + lane * 8;
#pragma unroll
      for (int nt = 0; nt < 4; ++nt) xnext[nt] = *(const v8f*)(xb + nt * 256);
    }

    // ---- layer 0: h0 = tanh(xw0[t] + h0 @ Whh0.T)   (2-deep WMMA chain)
#pragma unroll
    for (int nt = 0; nt < 4; ++nt) {
      CDu c;
      c.v = wmma_bf16(a0k0, Bh0[0][nt], xcur[nt]);
      c.v = wmma_bf16(a0k1, Bh0[1][nt], c.v);
#pragma unroll
      for (int v = 0; v < 8; ++v) c.e[v] = fast_tanh(c.e[v]);
      hn0[nt] = c.v;
    }
    // D-layout -> LDS (row-major) for transpose to A-layout
#pragma unroll
    for (int nt = 0; nt < 4; ++nt) {
      CDu c; c.v = hn0[nt];
#pragma unroll
      for (int v = 0; v < 8; ++v)
        hs0[(v + 8 * hi) * LDSS + nt * 16 + lm] = c.e[v];
    }
    v16bf a1k0 = load_afrag(hs0 + lm * LDSS, 0, hi);
    v16bf a1k1 = load_afrag(hs0 + lm * LDSS, 1, hi);

    // ---- layer 1: h1 = tanh(y1 @ Wih1.T + b1 + h1 @ Whh1.T)
    // two independent 2-deep chains + one vector add (shorter latency tail)
#pragma unroll
    for (int nt = 0; nt < 4; ++nt) {
      v8f ca = wmma_bf16(a1k0, Bi1[0][nt], biasC1[nt]);
      ca = wmma_bf16(a1k1, Bi1[1][nt], ca);
      v8f cb = wmma_bf16(ahk0, Bh1[0][nt], zeroC);
      cb = wmma_bf16(ahk1, Bh1[1][nt], cb);
      CDu c;
      c.v = ca + cb;
#pragma unroll
      for (int v = 0; v < 8; ++v) c.e[v] = fast_tanh(c.e[v]);
      hn1[nt] = c.v;
    }
#pragma unroll
    for (int nt = 0; nt < 4; ++nt) {
      CDu c; c.v = hn1[nt];
#pragma unroll
      for (int v = 0; v < 8; ++v)
        hs1[(v + 8 * hi) * LDSS + nt * 16 + lm] = c.e[v];
    }
    v16bf a2k0 = load_afrag(hs1 + lm * LDSS, 0, hi);
    v16bf a2k1 = load_afrag(hs1 + lm * LDSS, 1, hi);

    // ---- fused FC: out[b,t,:] = y2 @ Wfc.T + bfc   (row index = b*T + t)
#pragma unroll
    for (int nt = 0; nt < 4; ++nt) {
      CDu c;
      c.v = wmma_bf16(a2k0, Bfc_[0][nt], biasCf[nt]);
      c.v = wmma_bf16(a2k1, Bfc_[1][nt], c.v);
#pragma unroll
      for (int v = 0; v < 8; ++v)
        __builtin_nontemporal_store(
            c.e[v],
            &out[((size_t)(b0 + v + 8 * hi) * Tn + t) * Cn + nt * 16 + lm]);
    }

    // rotate fragments: h(t) frags already in registers for step t+1
    a0k0 = a1k0; a0k1 = a1k1;
    ahk0 = a2k0; ahk1 = a2k1;
#pragma unroll
    for (int nt = 0; nt < 4; ++nt) xcur[nt] = xnext[nt];
  }

  // final hidden states -> hout [L,B,H]
#pragma unroll
  for (int nt = 0; nt < 4; ++nt) {
    CDu c0; c0.v = hn0[nt];
    CDu c1; c1.v = hn1[nt];
#pragma unroll
    for (int v = 0; v < 8; ++v) {
      int m = b0 + v + 8 * hi;
      hout[(size_t)m * Hn + nt * 16 + lm] = c0.e[v];
      hout[(size_t)Bn * Hn + (size_t)m * Hn + nt * 16 + lm] = c1.e[v];
    }
  }
}

extern "C" void kernel_launch(void* const* d_in, const int* in_sizes, int n_in,
                              void* d_out, int out_size, void* d_ws,
                              size_t ws_size, hipStream_t stream) {
  (void)in_sizes; (void)n_in; (void)out_size; (void)ws_size;
  const float* x    = (const float*)d_in[0];
  const float* h0   = (const float*)d_in[1];
  const float* Wih0 = (const float*)d_in[2];
  const float* Whh0 = (const float*)d_in[3];
  const float* bih0 = (const float*)d_in[4];
  const float* bhh0 = (const float*)d_in[5];
  const float* Wih1 = (const float*)d_in[6];
  const float* Whh1 = (const float*)d_in[7];
  const float* bih1 = (const float*)d_in[8];
  const float* bhh1 = (const float*)d_in[9];
  const float* Wfc  = (const float*)d_in[10];
  const float* bfc  = (const float*)d_in[11];

  float* out  = (float*)d_out;
  float* hout = out + (size_t)Bn * Tn * Cn;
  float* xw0  = (float*)d_ws;  // T*B*H floats = 128 MB fragment-major scratch

  rnn_proj0_kernel<<<dim3(Tn * (Bn / 16) / 8), dim3(256), 0, stream>>>(
      x, Wih0, bih0, bhh0, xw0);
  rnn_scan_fused_kernel<<<dim3(Bn / 16), dim3(32), 0, stream>>>(
      xw0, h0, Whh0, Wih1, Whh1, bih1, bhh1, Wfc, bfc, out, hout);
}
